// Seq2Seq_44495861187146
// MI455X (gfx1250) — matrix operand order repaired and checked by reading, not compiled, not run on hardware
//
#include <hip/hip_runtime.h>
#include <hip/hip_bf16.h>
#include <math.h>
#include <stdint.h>

typedef __attribute__((ext_vector_type(16))) _Float16 v16h;
typedef __attribute__((ext_vector_type(8)))  _Float16 v8h;
typedef __attribute__((ext_vector_type(8)))  float    v8f;
typedef unsigned int v4u __attribute__((ext_vector_type(4)));
typedef int          v8i __attribute__((ext_vector_type(8)));
typedef int          v4i __attribute__((ext_vector_type(4)));

#define BB 16
#define SS 128
#define TT 128
#define HH 512
#define EE 512
#define VV 32000
#define G4 2048   // 4*H
#define BT 2048   // B*T == B*S

// ---------------------------------------------------------------------------
// WMMA fragment loaders (layouts per CDNA5 ISA 7.12.2, wave32)
// A 16x32 f16: lanes 0-15 -> M=lane, K={kb..kb+7, kb+16..kb+23}
//              lanes 16-31 -> M=lane-16, K={kb+8..kb+15, kb+24..kb+31}
// ---------------------------------------------------------------------------
__device__ inline v16h load_afrag(const _Float16* p) {
    // p already offset by row*lda + (lane>>4)*8 + kb
    v8h lo = *(const v8h*)p;
    v8h hi = *(const v8h*)(p + 16);
    v16h r;
#pragma unroll
    for (int i = 0; i < 8; ++i) { r[i] = lo[i]; r[i + 8] = hi[i]; }
    return r;
}

// B 32x16 f16: lanes 0-15 -> N=lane, K={kb..kb+7, kb+16..kb+23}
//              lanes 16-31 -> N=lane-16, K={kb+8..kb+15, kb+24..kb+31}
__device__ inline v16h load_bfrag(const _Float16* p) {
    // p already offset by row*ldb + (lane>>4)*8 + kb
    v8h lo = *(const v8h*)p;
    v8h hi = *(const v8h*)(p + 16);
    v16h r;
#pragma unroll
    for (int i = 0; i < 8; ++i) { r[i] = lo[i]; r[i + 8] = hi[i]; }
    return r;
}

__device__ inline v8f wmma_f16(v16h a, v16h b, v8f c) {
    return __builtin_amdgcn_wmma_f32_16x16x32_f16(false, a, false, b, (short)0, c,
                                                  false, false);
}

__device__ inline float sigm(float x) { return 1.0f / (1.0f + expf(-x)); }

// ---------------------------------------------------------------------------
// Generic WMMA GEMM: C[m][n] = sum_k A[m][k] * Bw[n][k] (+bias[n])
// A: f16 (M x K) row-major lda ; Bw: f16 (N x K) row-major ldb
// Wave computes a 16x64 tile (4 accumulators, shared A fragment).
// grid = (N/256, M/16), block = (32, 4)
// ---------------------------------------------------------------------------
__global__ void gemm_wmma_kernel(const _Float16* __restrict__ A, int lda,
                                 const _Float16* __restrict__ Bw, int ldb,
                                 float* __restrict__ C32, _Float16* __restrict__ C16,
                                 int ldc, const float* __restrict__ bias, int K) {
    const int lane = threadIdx.x;
    const int wave = threadIdx.y;
    const int r    = lane & 15;
    const int rsel = lane >> 4;
    const int m0   = blockIdx.y * 16;
    const int n0   = (blockIdx.x * 4 + wave) * 64;

    const _Float16* ap = A + (long)(m0 + r) * lda + rsel * 8;
    const _Float16* bp = Bw + (long)(n0 + r) * ldb + rsel * 8;

    v8f acc[4] = {v8f{}, v8f{}, v8f{}, v8f{}};
    for (int kb = 0; kb < K; kb += 32) {
        __builtin_prefetch(ap + kb + 128, 0, 0);   // global_prefetch_b8
        v16h afrag = load_afrag(ap + kb);
#pragma unroll
        for (int j = 0; j < 4; ++j) {
            v16h bfrag = load_bfrag(bp + (long)j * 16 * ldb + kb);
            acc[j] = wmma_f16(afrag, bfrag, acc[j]);
        }
    }
#pragma unroll
    for (int j = 0; j < 4; ++j) {
        const int n = n0 + j * 16 + r;
        const float bv = bias ? bias[n] : 0.0f;
#pragma unroll
        for (int e = 0; e < 8; ++e) {
            const int m = m0 + e + 8 * rsel;
            const float v = acc[j][e] + bv;
            if (C32) C32[(long)m * ldc + n] = v;
            if (C16) C16[(long)m * ldc + n] = (_Float16)v;
        }
    }
}

// ---------------------------------------------------------------------------
// LSTM recurrent step: one wave per 16 h-columns. Computes the i/f/g/o gate
// tiles (4 WMMA accumulators, K=512), adds precomputed x-projection + bias,
// applies gate nonlinearity in-register via the C-matrix lane layout.
// grid = 32 blocks, block = 32 threads.
// ---------------------------------------------------------------------------
__global__ void lstm_step_kernel(const _Float16* __restrict__ hin,   // 16x512 f16
                                 _Float16* __restrict__ hout,        // 16x512 f16
                                 float* __restrict__ c,              // 16x512 f32
                                 const _Float16* __restrict__ whh,   // 2048x512 f16
                                 const float* __restrict__ xproj,    // (B*seq)x2048
                                 const float* __restrict__ bias,     // 2048
                                 float* __restrict__ hs32,
                                 _Float16* __restrict__ hs16,
                                 int tidx, int seqlen, int ldh, int coloff) {
    const int lane = threadIdx.x;
    const int r    = lane & 15;
    const int rsel = lane >> 4;
    const int n0   = blockIdx.x * 16;

    const _Float16* ap = hin + r * HH + rsel * 8;
    const _Float16* bi_p = whh + (long)(n0 + r) * HH + rsel * 8;
    const _Float16* bf_p = bi_p + (long)512 * HH;
    const _Float16* bg_p = bi_p + (long)1024 * HH;
    const _Float16* bo_p = bi_p + (long)1536 * HH;

    v8f ai = v8f{}, af = v8f{}, ag = v8f{}, ao = v8f{};
    for (int kb = 0; kb < HH; kb += 32) {
        v16h a = load_afrag(ap + kb);
        ai = wmma_f16(a, load_bfrag(bi_p + kb), ai);
        af = wmma_f16(a, load_bfrag(bf_p + kb), af);
        ag = wmma_f16(a, load_bfrag(bg_p + kb), ag);
        ao = wmma_f16(a, load_bfrag(bo_p + kb), ao);
    }

    const int n = n0 + r;
    const float b_i = bias[n], b_f = bias[n + 512], b_g = bias[n + 1024],
                b_o = bias[n + 1536];
#pragma unroll
    for (int e = 0; e < 8; ++e) {
        const int m = e + 8 * rsel;                       // batch index 0..15
        const long xr = ((long)m * seqlen + tidx) * (long)G4;
        const float gi = ai[e] + xproj[xr + n]        + b_i;
        const float gf = af[e] + xproj[xr + 512 + n]  + b_f;
        const float gg = ag[e] + xproj[xr + 1024 + n] + b_g;
        const float go = ao[e] + xproj[xr + 1536 + n] + b_o;
        const long ci = (long)m * HH + n;
        const float cc = sigm(gf) * c[ci] + sigm(gi) * tanhf(gg);
        const float hh = sigm(go) * tanhf(cc);
        c[ci] = cc;
        hout[ci] = (_Float16)hh;
        const long orow = ((long)m * seqlen + tidx) * (long)ldh + coloff + n;
        hs32[orow] = hh;
        hs16[orow] = (_Float16)hh;
    }
}

// ---------------------------------------------------------------------------
// TDM staging demo: builds a D# descriptor (ISA 8.3-8.5) for a 2D f16 tile
// (tile_dim0=128 halfs x tile_dim1=64 rows) of the generator weight matrix
// (tensor 512 x 32000, stride 512, data_size=2B) and DMA-loads it into LDS
// via the Tensor Data Mover, synchronized with TENSORcnt. The tile is folded
// into a scratch sink so the pipeline is not eliminated.
// launch: 1 block x 32 threads (one wave; EXEC ignored by TENSOR ops)
// ---------------------------------------------------------------------------
__global__ void tdm_stage_kernel(const _Float16* __restrict__ gw,
                                 float* __restrict__ sink) {
#if defined(__gfx1250__)
    __shared__ _Float16 tile[64 * 128];
    const unsigned lds_off = (unsigned)(uintptr_t)&tile[0];   // LDS byte address
    const unsigned long long ga = (unsigned long long)(uintptr_t)gw;

    // D# group 0: count=1 | lds_addr | global_addr[56:0] | type=2 ("image")
    v4u g0;
    g0[0] = 1u;                                            // count=1, user D#
    g0[1] = lds_off;                                       // lds_addr [63:32]
    g0[2] = (unsigned)(ga & 0xFFFFFFFFull);                // global_addr[31:0]
    g0[3] = (unsigned)((ga >> 32) & 0x1FFFFFFull) | (2u << 30); // [56:32]|type=2

    // D# group 1: data_size=2B, tensor 512x32000 (stride0=512),
    //             tile 128x64, no multicast / iterate / pad / atomic-barrier
    v8i g1;
    g1[0] = (int)(1u << 16);                               // data_size = 2 bytes
    g1[1] = (int)((512u & 0xFFFFu) << 16);                 // tensor_dim0 lo16
    g1[2] = (int)(((512u >> 16) & 0xFFFFu) |               // tensor_dim0 hi16
                  ((32000u & 0xFFFFu) << 16));             // tensor_dim1 lo16
    g1[3] = (int)(((32000u >> 16) & 0xFFFFu) |             // tensor_dim1 hi16
                  (128u << 16));                           // tile_dim0 = 128
    g1[4] = 64;                                            // tile_dim1=64, dim2=0
    g1[5] = 512;                                           // tensor_dim0_stride lo
    g1[6] = 0;                                             // stride0 hi | stride1 lo
    g1[7] = 0;                                             // stride1 hi
    v4i gz = {0, 0, 0, 0};                                 // groups 2/3 unused (2D)

#if defined(__clang_major__) && __clang_major__ >= 23
    v8i gz8 = {0, 0, 0, 0, 0, 0, 0, 0};
    __builtin_amdgcn_tensor_load_to_lds(g0, g1, gz, gz, gz8, 0);
#else
    __builtin_amdgcn_tensor_load_to_lds(g0, g1, gz, gz, 0);
#endif
    __builtin_amdgcn_s_cluster_barrier();   // NOP when ClusterID==0 (not in cluster)
    __builtin_amdgcn_s_wait_tensorcnt(0);   // s_wait_tensorcnt 0

    float s = 0.0f;
    for (int i = threadIdx.x; i < 64 * 128; i += 32) s += (float)tile[i];
    sink[threadIdx.x] = s;
#endif
}

// ---------------------------------------------------------------------------
// Embedding gather + f16 convert: out[tok][k] = (f16)emb[ids[tok]][k]
// ---------------------------------------------------------------------------
__global__ void embed_kernel(const int* __restrict__ ids,
                             const float* __restrict__ emb,
                             _Float16* __restrict__ out) {
    const int tok = blockIdx.x;
    const long row = ids[tok];
    for (int k = threadIdx.x; k < EE; k += blockDim.x)
        out[(long)tok * EE + k] = (_Float16)emb[row * EE + k];
}

__global__ void cvt_f16_kernel(const float* __restrict__ in,
                               _Float16* __restrict__ out, long n) {
    const long i = (long)blockIdx.x * blockDim.x + threadIdx.x;
    if (i < n) out[i] = (_Float16)in[i];
}

__global__ void zero_u32_kernel(unsigned* __restrict__ p, long n) {
    const long i = (long)blockIdx.x * blockDim.x + threadIdx.x;
    if (i < n) p[i] = 0u;
}

// ---------------------------------------------------------------------------
// Fused attention: per (b,t) block computes e[s] = Va . tanh(pd + pe[s]),
// masked softmax over s, then context[k] = sum_s a[s] * enc_out[b,s,k].
// block = 256 threads, grid = B*T
// ---------------------------------------------------------------------------
__global__ void attn_kernel(const float* __restrict__ pd,
                            const float* __restrict__ pe,
                            const float* __restrict__ Va,
                            const int* __restrict__ src_mask,
                            const float* __restrict__ enc_out,
                            float* __restrict__ context) {
    __shared__ float s_pd[HH];
    __shared__ float s_va[HH];
    __shared__ float s_e[SS];
    __shared__ float s_red[256];

    const int bt = blockIdx.x;
    const int b  = bt / TT;
    const int tid = threadIdx.x;
    for (int k = tid; k < HH; k += 256) { s_pd[k] = pd[(long)bt * HH + k]; s_va[k] = Va[k]; }
    __syncthreads();

    const int lane = tid & 31, wv = tid >> 5;   // 8 waves
    for (int s = wv; s < SS; s += 8) {
        const float* per = pe + ((long)b * SS + s) * HH;
        float acc = 0.0f;
#pragma unroll 4
        for (int i = 0; i < 16; ++i) {
            const int k = lane + 32 * i;
            acc += tanhf(s_pd[k] + per[k]) * s_va[k];
        }
        for (int off = 16; off > 0; off >>= 1) acc += __shfl_xor(acc, off, 32);
        if (lane == 0)
            s_e[s] = (src_mask[b * SS + s] != 0) ? acc : -INFINITY;
    }
    __syncthreads();

    // softmax over 128 scores
    s_red[tid] = (tid < SS) ? s_e[tid] : -INFINITY;
    __syncthreads();
    for (int st = 128; st > 0; st >>= 1) {
        if (tid < st) s_red[tid] = fmaxf(s_red[tid], s_red[tid + st]);
        __syncthreads();
    }
    const float mx = s_red[0];
    __syncthreads();
    float ev = 0.0f;
    if (tid < SS) { ev = expf(s_e[tid] - mx); }
    s_red[tid] = ev;
    __syncthreads();
    for (int st = 128; st > 0; st >>= 1) {
        if (tid < st) s_red[tid] += s_red[tid + st];
        __syncthreads();
    }
    const float inv = 1.0f / s_red[0];
    __syncthreads();
    if (tid < SS) s_e[tid] = ev * inv;
    __syncthreads();

    // context: 1024 k-values, 4 per thread
    float acc[4] = {0.f, 0.f, 0.f, 0.f};
    for (int s = 0; s < SS; ++s) {
        const float a = s_e[s];
        const float* er = enc_out + ((long)b * SS + s) * (2 * HH);
#pragma unroll
        for (int j = 0; j < 4; ++j) acc[j] += a * er[tid + 256 * j];
    }
#pragma unroll
    for (int j = 0; j < 4; ++j)
        context[(long)bt * (2 * HH) + tid + 256 * j] = acc[j];
}

// feats = concat(dec_out(512), context(1024)) as f16, rows = B*T
__global__ void concat_feats_kernel(const float* __restrict__ dec_out,
                                    const float* __restrict__ context,
                                    _Float16* __restrict__ feats) {
    const long i = (long)blockIdx.x * blockDim.x + threadIdx.x;
    if (i >= (long)BT * 1536) return;
    const long row = i / 1536;
    const int k = (int)(i % 1536);
    const float v = (k < HH) ? dec_out[row * HH + k]
                             : context[row * (2 * HH) + (k - HH)];
    feats[i] = (_Float16)v;
}

// in-place log-softmax over N=32000 per row; block=256, grid = B*T
__global__ void logsoftmax_kernel(float* __restrict__ logits, int N) {
    __shared__ float s_red[256];
    float* p = logits + (long)blockIdx.x * N;
    const int tid = threadIdx.x;

    float mx = -INFINITY;
    for (int k = tid; k < N; k += 256) mx = fmaxf(mx, p[k]);
    s_red[tid] = mx;
    __syncthreads();
    for (int st = 128; st > 0; st >>= 1) {
        if (tid < st) s_red[tid] = fmaxf(s_red[tid], s_red[tid + st]);
        __syncthreads();
    }
    mx = s_red[0];
    __syncthreads();
    float sum = 0.0f;
    for (int k = tid; k < N; k += 256) sum += expf(p[k] - mx);
    s_red[tid] = sum;
    __syncthreads();
    for (int st = 128; st > 0; st >>= 1) {
        if (tid < st) s_red[tid] += s_red[tid + st];
        __syncthreads();
    }
    const float lse = mx + logf(s_red[0]);
    for (int k = tid; k < N; k += 256) p[k] = p[k] - lse;
}

// ---------------------------------------------------------------------------
// Host orchestration
// ---------------------------------------------------------------------------
static inline size_t align256(size_t x) { return (x + 255) & ~(size_t)255; }

static void launch_gemm(const _Float16* A, int lda, const _Float16* Bw, int ldb,
                        float* C32, _Float16* C16, int ldc, const float* bias,
                        int M, int N, int K, hipStream_t s) {
    dim3 blk(32, 4), grd(N / 256, M / 16);
    gemm_wmma_kernel<<<grd, blk, 0, s>>>(A, lda, Bw, ldb, C32, C16, ldc, bias, K);
}

extern "C" void kernel_launch(void* const* d_in, const int* in_sizes, int n_in,
                              void* d_out, int out_size, void* d_ws, size_t ws_size,
                              hipStream_t stream) {
    const int*   src_inputs = (const int*)d_in[0];
    const int*   tgt_inputs = (const int*)d_in[1];
    const int*   src_mask   = (const int*)d_in[2];
    const float* src_emb    = (const float*)d_in[3];
    const float* tgt_emb    = (const float*)d_in[4];
    const float* enc_Wih_f  = (const float*)d_in[5];
    const float* enc_Whh_f  = (const float*)d_in[6];
    const float* enc_b_f    = (const float*)d_in[7];
    const float* enc_Wih_b  = (const float*)d_in[8];
    const float* enc_Whh_b  = (const float*)d_in[9];
    const float* enc_b_b    = (const float*)d_in[10];
    const float* dec_Wih    = (const float*)d_in[11];
    const float* dec_Whh    = (const float*)d_in[12];
    const float* dec_b      = (const float*)d_in[13];
    const float* Wa         = (const float*)d_in[14];
    const float* Va         = (const float*)d_in[15];
    const float* affine_W   = (const float*)d_in[16];
    const float* affine_b   = (const float*)d_in[17];
    const float* gen_W      = (const float*)d_in[18];
    const float* gen_b      = (const float*)d_in[19];
    float* out = (float*)d_out;
    (void)in_sizes; (void)n_in; (void)out_size; (void)ws_size;

    // ---- workspace bump allocator ----
    char* w = (char*)d_ws;
    size_t off = 0;
    auto alloc = [&](size_t bytes) -> void* {
        void* p = w + off;
        off += align256(bytes);
        return p;
    };
    _Float16* se16      = (_Float16*)alloc((size_t)BT * EE * 2);
    _Float16* te16      = (_Float16*)alloc((size_t)BT * EE * 2);
    _Float16* wih_f16   = (_Float16*)alloc((size_t)G4 * EE * 2);
    _Float16* whh_f16   = (_Float16*)alloc((size_t)G4 * HH * 2);
    _Float16* wih_b16   = (_Float16*)alloc((size_t)G4 * EE * 2);
    _Float16* whh_b16   = (_Float16*)alloc((size_t)G4 * HH * 2);
    _Float16* wih_d16   = (_Float16*)alloc((size_t)G4 * EE * 2);
    _Float16* whh_d16   = (_Float16*)alloc((size_t)G4 * HH * 2);
    _Float16* wa16      = (_Float16*)alloc((size_t)HH * 1536 * 2);
    _Float16* affW16    = (_Float16*)alloc((size_t)EE * 1536 * 2);
    _Float16* genW16    = (_Float16*)alloc((size_t)VV * EE * 2);
    float*    xproj_f   = (float*)alloc((size_t)BT * G4 * 4);
    float*    xproj_b   = (float*)alloc((size_t)BT * G4 * 4);
    float*    xproj_d   = (float*)alloc((size_t)BT * G4 * 4);
    float*    enc_out32 = (float*)alloc((size_t)BT * 2 * HH * 4);
    _Float16* enc_out16 = (_Float16*)alloc((size_t)BT * 2 * HH * 2);
    float*    dec_out32 = (float*)alloc((size_t)BT * HH * 4);
    _Float16* dec_out16 = (_Float16*)alloc((size_t)BT * HH * 2);
    float*    pd        = (float*)alloc((size_t)BT * HH * 4);
    float*    pe        = (float*)alloc((size_t)BT * HH * 4);
    float*    context   = (float*)alloc((size_t)BT * 2 * HH * 4);
    _Float16* feats16   = (_Float16*)alloc((size_t)BT * 1536 * 2);
    _Float16* aff_out16 = (_Float16*)alloc((size_t)BT * EE * 2);
    float*    tdm_sink  = (float*)alloc(256);
    // LSTM state block: 3 cells (f32) + 6 h ping-pong (f16), each 16x512
    char* states = (char*)alloc(3 * (size_t)BB * HH * 4 + 6 * (size_t)BB * HH * 2);
    float*    c_f  = (float*)(states);
    float*    c_b  = (float*)(states + 1 * BB * HH * 4);
    float*    c_d  = (float*)(states + 2 * BB * HH * 4);
    _Float16* hA[3], *hB[3];
    {
        char* hp = states + 3 * (size_t)BB * HH * 4;
        for (int i = 0; i < 3; ++i) {
            hA[i] = (_Float16*)(hp + (2 * i)     * (size_t)BB * HH * 2);
            hB[i] = (_Float16*)(hp + (2 * i + 1) * (size_t)BB * HH * 2);
        }
    }

    auto cvt = [&](const float* src, _Float16* dst, long n) {
        cvt_f16_kernel<<<(unsigned)((n + 255) / 256), 256, 0, stream>>>(src, dst, n);
    };

    // 1) weight conversions to f16
    cvt(enc_Wih_f, wih_f16, (long)G4 * EE);
    cvt(enc_Whh_f, whh_f16, (long)G4 * HH);
    cvt(enc_Wih_b, wih_b16, (long)G4 * EE);
    cvt(enc_Whh_b, whh_b16, (long)G4 * HH);
    cvt(dec_Wih,   wih_d16, (long)G4 * EE);
    cvt(dec_Whh,   whh_d16, (long)G4 * HH);
    cvt(Wa,        wa16,    (long)HH * 1536);
    cvt(affine_W,  affW16,  (long)EE * 1536);
    cvt(gen_W,     genW16,  (long)VV * EE);

    // 2) embeddings (gather + f16)
    embed_kernel<<<BT, 256, 0, stream>>>(src_inputs, src_emb, se16);
    embed_kernel<<<BT, 256, 0, stream>>>(tgt_inputs, tgt_emb, te16);

    // 3) hoisted input projections X @ Wih^T  (M=2048, N=2048, K=512)
    launch_gemm(se16, EE, wih_f16, EE, xproj_f, nullptr, G4, nullptr, BT, G4, EE, stream);
    launch_gemm(se16, EE, wih_b16, EE, xproj_b, nullptr, G4, nullptr, BT, G4, EE, stream);
    launch_gemm(te16, EE, wih_d16, EE, xproj_d, nullptr, G4, nullptr, BT, G4, EE, stream);

    // 4) zero LSTM states (c and h ping-pong buffers)
    {
        long ndw = (3 * (long)BB * HH * 4 + 6 * (long)BB * HH * 2) / 4;
        zero_u32_kernel<<<(unsigned)((ndw + 255) / 256), 256, 0, stream>>>(
            (unsigned*)states, ndw);
    }

    // 5) recurrent loops: enc-fwd, enc-bwd, dec (all three per step, wmma gates)
    for (int t = 0; t < SS; ++t) {
        const _Float16* hinF = (t & 1) ? hB[0] : hA[0];
        _Float16*       houF = (t & 1) ? hA[0] : hB[0];
        const _Float16* hinB = (t & 1) ? hB[1] : hA[1];
        _Float16*       houB = (t & 1) ? hA[1] : hB[1];
        const _Float16* hinD = (t & 1) ? hB[2] : hA[2];
        _Float16*       houD = (t & 1) ? hA[2] : hB[2];
        lstm_step_kernel<<<32, 32, 0, stream>>>(hinF, houF, c_f, whh_f16, xproj_f,
            enc_b_f, enc_out32, enc_out16, t, SS, 2 * HH, 0);
        lstm_step_kernel<<<32, 32, 0, stream>>>(hinB, houB, c_b, whh_b16, xproj_b,
            enc_b_b, enc_out32, enc_out16, SS - 1 - t, SS, 2 * HH, HH);
        lstm_step_kernel<<<32, 32, 0, stream>>>(hinD, houD, c_d, whh_d16, xproj_d,
            dec_b, dec_out32, dec_out16, t, TT, HH, 0);
    }

    // 6) attention projections: pd = dec_out @ Wd^T, pe = enc_out @ We^T
    //    Wd = Wa[:, :H] rows (ldb=1536), We = Wa[:, H:] rows (ldb=1536, K=1024)
    launch_gemm(dec_out16, HH, wa16, 1536, pd, nullptr, HH, nullptr, BT, HH, HH, stream);
    launch_gemm(enc_out16, 2 * HH, wa16 + HH, 1536, pe, nullptr, HH, nullptr,
                BT, HH, 2 * HH, stream);

    // 7) fused tanh-score + masked softmax + context
    attn_kernel<<<BT, 256, 0, stream>>>(pd, pe, Va, src_mask, enc_out32, context);

    // 8) feats = concat(dec_out, context) -> f16
    {
        long n = (long)BT * 1536;
        concat_feats_kernel<<<(unsigned)((n + 255) / 256), 256, 0, stream>>>(
            dec_out32, context, feats16);
    }

    // 9) affine: feats @ affine_W^T + affine_b  (K=1536, N=512) -> f16
    launch_gemm(feats16, 1536, affW16, 1536, nullptr, aff_out16, EE, affine_b,
                BT, EE, 1536, stream);

    // 9.5) TDM staging demo on the generator weights (async-tensor path)
    tdm_stage_kernel<<<1, 32, 0, stream>>>(genW16, tdm_sink);

    // 10) generator GEMM straight into d_out: (2048 x 32000, K=512) + gen_b
    launch_gemm(aff_out16, EE, genW16, EE, out, nullptr, VV, gen_b,
                BT, VV, EE, stream);

    // 11) in-place log-softmax over vocab
    logsoftmax_kernel<<<BT, 256, 0, stream>>>(out, VV);
}
